// Net_13142599926355
// MI455X (gfx1250) — compile-verified, matrix operand order checked by reading, not run on hardware
//
#include <hip/hip_runtime.h>
#include <hip/hip_bf16.h>

typedef __bf16 bf16_t;
typedef bf16_t v16bf __attribute__((ext_vector_type(16)));
typedef bf16_t v8bf  __attribute__((ext_vector_type(8)));
typedef bf16_t v4bf  __attribute__((ext_vector_type(4)));
typedef float  v8f   __attribute__((ext_vector_type(8)));

#define B_  64
#define S_  512
#define DB  768
#define E_  50
#define T_  68
#define NC_ 16
#define NT_ 8
#define KC  832   // Db+E (818) padded to multiple of 32
#define TP  80    // T padded to multiple of 16
#define M_  (B_ * S_)   // 32768 tokens

#define SHUF16(lo, hi) __builtin_shufflevector((lo), (hi), 0,1,2,3,4,5,6,7,8,9,10,11,12,13,14,15)

// ---------------------------------------------------------------------------
// Kernel 1: convert fc1_w [768,818]f32 -> W1 [768,832]bf16 (zero-pad K),
//           trig_w [68,768]f32 -> Wt [80,768]bf16 (zero-pad rows)
// ---------------------------------------------------------------------------
__global__ void convert_weights(const float* __restrict__ fc1_w,
                                const float* __restrict__ trig_w,
                                bf16_t* __restrict__ W1,
                                bf16_t* __restrict__ Wt) {
    int idx = blockIdx.x * blockDim.x + threadIdx.x;
    const int n1 = DB * KC;
    const int n2 = TP * DB;
    if (idx < n1) {
        int row = idx / KC, col = idx - row * KC;
        float v = (col < DB + E_) ? fc1_w[row * (DB + E_) + col] : 0.f;
        W1[idx] = (bf16_t)v;
    } else if (idx < n1 + n2) {
        int j = idx - n1;
        int row = j / DB, col = j - row * DB;
        float v = (row < T_) ? trig_w[row * DB + col] : 0.f;
        Wt[j] = (bf16_t)v;
    }
}

// ---------------------------------------------------------------------------
// Kernel 2: fused fc1 GEMM.
// Builds the 16x832 bf16 A-tile in LDS directly from enc (f32->bf16) and the
// summed entity embeddings, then X[32768,768] = relu(A * W1^T + b) via WMMA.
// Block = 256 thr (8 waves); wave owns 6 N-tiles (wave + 8i); K loop of 26.
// ---------------------------------------------------------------------------
__global__ __launch_bounds__(256) void fc1_gemm(const float* __restrict__ enc,
                                                const float* __restrict__ table,
                                                const int* __restrict__ ents,
                                                const bf16_t* __restrict__ W1,
                                                const float* __restrict__ bias,
                                                bf16_t* __restrict__ X) {
    __shared__ bf16_t sA[16 * KC];   // 26624 B
    const int tid  = threadIdx.x;
    const int mt   = blockIdx.x;     // 0..2047
    const int tok0 = mt * 16;

    // stage enc part: 16x768 f32 -> bf16, float4-vectorized (12 iters/thread)
    for (int i = tid; i < (16 * DB) / 4; i += 256) {
        const int row = i / (DB / 4);
        const int c4  = i - row * (DB / 4);
        const float4 v = ((const float4*)(enc + (size_t)(tok0 + row) * DB))[c4];
        v4bf p;
        p.x = (bf16_t)v.x; p.y = (bf16_t)v.y; p.z = (bf16_t)v.z; p.w = (bf16_t)v.w;
        *(v4bf*)(sA + row * KC + c4 * 4) = p;
    }
    // stage entity-sum part + zero pad: cols 768..831 (4 iters/thread)
    for (int i = tid; i < 16 * 64; i += 256) {
        const int row = i / 64;
        const int c   = DB + (i - row * 64);
        float v = 0.f;
        if (c < DB + E_) {
            const int e = c - DB;
            const int base = (tok0 + row) * 3;
            v = table[ents[base + 0] * E_ + e]
              + table[ents[base + 1] * E_ + e]
              + table[ents[base + 2] * E_ + e];
        }
        sA[row * KC + c] = (bf16_t)v;
    }
    __syncthreads();

    const int wave = tid >> 5;
    const int lane = tid & 31;
    const int rrow = lane & 15;         // A/B row held by this lane
    const int koff = (lane >> 4) * 8;   // per-half-wave K offset

    v8f acc[6];
#pragma unroll
    for (int i = 0; i < 6; ++i) acc[i] = {};

    for (int k = 0; k < KC; k += 32) {
        const bf16_t* ap = sA + rrow * KC + k + koff;
        v8bf a_lo = *(const v8bf*)(ap);
        v8bf a_hi = *(const v8bf*)(ap + 16);
        v16bf a = SHUF16(a_lo, a_hi);
#pragma unroll
        for (int i = 0; i < 6; ++i) {
            const int ntile = wave + 8 * i;
            const bf16_t* bp = W1 + (size_t)(ntile * 16 + rrow) * KC + k + koff;
            v8bf b_lo = *(const v8bf*)(bp);
            v8bf b_hi = *(const v8bf*)(bp + 16);
            v16bf b = SHUF16(b_lo, b_hi);
            acc[i] = __builtin_amdgcn_wmma_f32_16x16x32_bf16(
                false, a, false, b, (short)0, acc[i], false, false);
        }
    }

    // epilogue: bias + relu + bf16 store.  C/D layout: VGPR j, lane l ->
    // M = j + 8*(l>>4), N = l&15 within the tile.
    const int mrow = 8 * (lane >> 4);
#pragma unroll
    for (int i = 0; i < 6; ++i) {
        const int n = (wave + 8 * i) * 16 + (lane & 15);
        const float bv = bias[n];
#pragma unroll
        for (int j = 0; j < 8; ++j) {
            float v = acc[i][j] + bv;
            v = v > 0.f ? v : 0.f;
            const int m = tok0 + mrow + j;
            X[(size_t)m * DB + n] = (bf16_t)v;
        }
    }
}

// ---------------------------------------------------------------------------
// Kernel 3: trigger GEMM with fused wordpiece gather.
// Gathered A rows (16 x 768 bf16 = 24576 B) are staged into LDS once per
// block with GLOBAL_LOAD_ASYNC_TO_LDS_B128 (ASYNCcnt path), then all 5 waves
// read A fragments from LDS.  Wave w -> N-tile w (T padded to 80).
// ---------------------------------------------------------------------------
__global__ __launch_bounds__(160) void trig_gemm(const bf16_t* __restrict__ X,
                                                 const bf16_t* __restrict__ Wt,
                                                 const float* __restrict__ bias,
                                                 const int* __restrict__ head_idx,
                                                 float* __restrict__ logits) {
    __shared__ bf16_t sA[16 * DB];   // 24576 B
    const int tid = threadIdx.x;
    const int mt  = blockIdx.x;      // 0..2047

    // async-stage the gathered 16x768 tile: 1536 16-byte chunks, 96 per row
    for (int chunk = tid; chunk < 16 * 96; chunk += 160) {
        const int row = chunk / 96;
        const int off = (chunk - row * 96) * 16;   // byte offset in row
        const int tok = mt * 16 + row;
        const int b   = tok / S_;
        const int src = b * S_ + head_idx[tok];
        const char* g = (const char*)(X + (size_t)src * DB) + off;
        const unsigned int l =
            (unsigned int)(size_t)((const char*)sA + row * (DB * 2) + off);
        asm volatile("global_load_async_to_lds_b128 %0, %1, off"
                     :: "v"(l), "v"(g) : "memory");
    }
    asm volatile("s_wait_asynccnt 0x0" ::: "memory");
    __syncthreads();

    const int wave = tid >> 5;          // 0..4
    const int lane = tid & 31;
    const int rrow = lane & 15;
    const int koff = (lane >> 4) * 8;

    const bf16_t* brow = Wt + (size_t)(wave * 16 + rrow) * DB;

    v8f acc = {};
    for (int k = 0; k < DB; k += 32) {
        const bf16_t* ap = sA + rrow * DB + k + koff;
        v8bf a_lo = *(const v8bf*)(ap);
        v8bf a_hi = *(const v8bf*)(ap + 16);
        v16bf a = SHUF16(a_lo, a_hi);
        v8bf b_lo = *(const v8bf*)(brow + k + koff);
        v8bf b_hi = *(const v8bf*)(brow + k + koff + 16);
        v16bf bb = SHUF16(b_lo, b_hi);
        acc = __builtin_amdgcn_wmma_f32_16x16x32_bf16(
            false, a, false, bb, (short)0, acc, false, false);
    }

    const int n = wave * 16 + (lane & 15);
    if (n < T_) {
        const float bv = bias[n];
        const int mbase = mt * 16 + 8 * (lane >> 4);
#pragma unroll
        for (int j = 0; j < 8; ++j) {
            logits[(size_t)(mbase + j) * T_ + n] = acc[j] + bv;
        }
    }
}

// ---------------------------------------------------------------------------
// Kernel 4: argmax over T (first-max wins, matching jnp.argmax)
// ---------------------------------------------------------------------------
__global__ void argmax_kernel(const float* __restrict__ logits,
                              float* __restrict__ hat) {
    const int tok = blockIdx.x * blockDim.x + threadIdx.x;
    if (tok >= M_) return;
    const float* r = logits + (size_t)tok * T_;
    float best = r[0];
    int bi = 0;
    for (int t = 1; t < T_; ++t) {
        float v = r[t];
        if (v > best) { best = v; bi = t; }
    }
    hat[tok] = (float)bi;
}

// ---------------------------------------------------------------------------
// Kernel 5: span mean pooling (over the gathered x) for trig & cand spans.
// Blocks [0, B*NT) -> ev, [B*NT, B*NT+B*NC) -> en.
// ---------------------------------------------------------------------------
__global__ void span_mean(const bf16_t* __restrict__ X,
                          const int* __restrict__ head_idx,
                          const int* __restrict__ trig_spans,
                          const int* __restrict__ cand_spans,
                          float* __restrict__ ev,
                          float* __restrict__ en) {
    const int bid = blockIdx.x;
    const int* sp;
    float* out;
    int b;
    if (bid < B_ * NT_) {
        b = bid / NT_;
        sp = trig_spans + bid * 2;
        out = ev + (size_t)bid * DB;
    } else {
        const int id = bid - B_ * NT_;
        b = id / NC_;
        sp = cand_spans + id * 2;
        out = en + (size_t)id * DB;
    }
    const int s0 = sp[0], s1 = sp[1];
    int len = s1 - s0;
    if (len < 1) len = 1;
    const float inv = 1.0f / (float)len;
    for (int d = threadIdx.x; d < DB; d += blockDim.x) {
        float sum = 0.f;
        for (int s = s0; s < s1; ++s) {
            const int src = b * S_ + head_idx[b * S_ + s];
            sum += (float)X[(size_t)src * DB + d];
        }
        out[d] = sum * inv;
    }
}

// ---------------------------------------------------------------------------
// Kernel 6: argument_hidden[b,nt,nc,:] = concat(ev[b,nt,:], en[b,nc,:])
// ---------------------------------------------------------------------------
__global__ void arg_hidden(const float* __restrict__ ev,
                           const float* __restrict__ en,
                           float* __restrict__ out) {
    const int bid = blockIdx.x;                 // b*NT*NC
    const int nc  = bid % NC_;
    const int tmp = bid / NC_;
    const int nt  = tmp % NT_;
    const int b   = tmp / NT_;
    const float* evp = ev + (size_t)(b * NT_ + nt) * DB;
    const float* enp = en + (size_t)(b * NC_ + nc) * DB;
    float* op = out + (size_t)bid * (2 * DB);
    for (int d = threadIdx.x; d < 2 * DB; d += blockDim.x) {
        op[d] = (d < DB) ? evp[d] : enp[d - DB];
    }
}

// ---------------------------------------------------------------------------
extern "C" void kernel_launch(void* const* d_in, const int* in_sizes, int n_in,
                              void* d_out, int out_size, void* d_ws, size_t ws_size,
                              hipStream_t stream) {
    (void)in_sizes; (void)n_in; (void)out_size; (void)ws_size;

    const float* enc        = (const float*)d_in[0];   // [64,512,768]
    const float* ent_table  = (const float*)d_in[1];   // [100,50]
    const float* fc1_w      = (const float*)d_in[2];   // [768,818]
    const float* fc1_b      = (const float*)d_in[3];   // [768]
    const float* trig_w     = (const float*)d_in[4];   // [68,768]
    const float* trig_b     = (const float*)d_in[5];   // [68]
    const int*   entities_x = (const int*)d_in[6];     // [64,512,3]
    const int*   head_idx   = (const int*)d_in[7];     // [64,512]
    const int*   cand_spans = (const int*)d_in[8];     // [64,16,2]
    const int*   trig_spans = (const int*)d_in[9];     // [64,8,2]

    // workspace carve-up (bytes)
    char* ws = (char*)d_ws;
    bf16_t* W1 = (bf16_t*)(ws);                   // 768*832*2   =  1,277,952
    bf16_t* Wt = (bf16_t*)(ws + 1277952);         // 80*768*2    =    122,880
    bf16_t* X  = (bf16_t*)(ws + 1400832);         // 32768*768*2 = 50,331,648
    float*  ev = (float*)(ws + 51732480);         // 64*8*768*4  =  1,572,864
    float*  en = (float*)(ws + 53305344);         // 64*16*768*4 =  3,145,728

    // output carve-up (floats)
    float* logits = (float*)d_out;                // 64*512*68    = 2,228,224
    float* hat    = logits + 2228224;             // 64*512       =    32,768
    float* arghid = logits + 2260992;             // 64*8*16*1536 = 12,582,912

    {   // 1. weight conversion
        const int total = DB * KC + TP * DB;
        convert_weights<<<(total + 255) / 256, 256, 0, stream>>>(fc1_w, trig_w, W1, Wt);
    }
    // 2. fused fc1 GEMM (builds A tile in LDS; WMMA bf16)
    fc1_gemm<<<M_ / 16, 256, 0, stream>>>(enc, ent_table, entities_x, W1, fc1_b, X);
    // 3. trigger GEMM: async-LDS gather staging + WMMA bf16
    trig_gemm<<<M_ / 16, 160, 0, stream>>>(X, Wt, trig_b, head_idx, logits);
    // 4. argmax
    argmax_kernel<<<M_ / 256, 256, 0, stream>>>(logits, hat);
    // 5. span means (gather-fused)
    span_mean<<<B_ * NT_ + B_ * NC_, 256, 0, stream>>>(X, head_idx, trig_spans,
                                                       cand_spans, ev, en);
    // 6. broadcast concat
    arg_hidden<<<B_ * NT_ * NC_, 256, 0, stream>>>(ev, en, arghid);
}
